// DynamicsModel_5068061409704
// MI455X (gfx1250) — matrix operand order repaired and checked by reading, not compiled, not run on hardware
//
#include <hip/hip_runtime.h>

// ---------------------------------------------------------------------------
// LSTM (B=256, T=256, F=128, H=1024, OUT=128) on gfx1250.
// Strategy: bf16 WMMA (v_wmma_f32_16x16x32_bf16, f32 accum) for all GEMMs.
//  - prep kernels: f32->bf16 conversion of x / W_ih / W_hh / W_fc / h0,
//    bias combine (b_ih + b_hh).
//  - 256 per-timestep kernels: fused [x_t|h] @ [W_ih|W_hh]^T for all 4 gates
//    + gate activations + c/h update. h double-buffered in bf16.
//  - final FC kernel: sigmoid(h) @ W_fc^T + b_fc via WMMA.
// ---------------------------------------------------------------------------

#define B_DIM   256
#define T_DIM   256
#define F_DIM   128
#define H_DIM   1024
#define OUT_DIM 128

typedef __attribute__((ext_vector_type(16))) __bf16 v16bf;
typedef __attribute__((ext_vector_type(8)))  float  v8f;

union FragU { uint4 u[2]; v16bf v; unsigned short s[16]; };

__device__ __forceinline__ float bf2f(unsigned short s) {
  unsigned u = ((unsigned)s) << 16;
  return __builtin_bit_cast(float, u);
}
__device__ __forceinline__ unsigned short f2bf(float f) {
  unsigned u = __builtin_bit_cast(unsigned, f);
  u += 0x7FFFu + ((u >> 16) & 1u);          // round-to-nearest-even
  return (unsigned short)(u >> 16);
}
__device__ __forceinline__ float sigmoidf_fast(float x) {
  return 1.0f / (1.0f + __expf(-x));
}
__device__ __forceinline__ float tanhf_fast(float x) {
  return 1.0f - 2.0f / (__expf(2.0f * x) + 1.0f);   // saturates to +/-1, no NaN
}

// A fragment: 16x32 bf16 tile, rows m_base..m_base+15, K cols kc..kc+31.
// Lane l holds row m_base+(l&15); lanes 0-15 K={kc..kc+7, kc+16..kc+23},
// lanes 16-31 K shifted by 8 (per ISA 16-bit A layout). Two 16B loads.
__device__ __forceinline__ v16bf load_frag_a(const unsigned short* __restrict__ base,
                                             int ld, int m_base, int kc, int lane) {
  const int row   = m_base + (lane & 15);
  const int khalf = (lane >> 4) << 3;                 // 0 or 8
  const unsigned short* p = base + (size_t)row * ld + kc + khalf;
  FragU f;
  f.u[0] = *(const uint4*)(p);                        // elems 0..7  (K = kc+khalf .. +7)
  f.u[1] = *(const uint4*)(p + 16);                   // elems 8..15 (K = kc+16+khalf ..)
  return f.v;
}

// B fragment: 32x16 bf16 tile (K x N); B[k][n] = W[n][k] for row-major W[N,K].
// Lane l holds column n_base+(l&15); lanes 0-15 K={kc..kc+7, kc+16..kc+23},
// lanes 16-31 K shifted by 8 (same order as A). Two 16B loads.
__device__ __forceinline__ v16bf load_frag_b(const unsigned short* __restrict__ base,
                                             int ld, int n_base, int kc, int lane) {
  const int col  = n_base + (lane & 15);
  const int koff = (lane >> 4) << 3;                  // 0 or 8
  const unsigned short* p = base + (size_t)col * ld + kc + koff;
  FragU f;
  f.u[0] = *(const uint4*)(p);
  f.u[1] = *(const uint4*)(p + 16);
  return f.v;
}

// ---------------------------------------------------------------------------
// Prep kernels
// ---------------------------------------------------------------------------
__global__ void cvt_bf16_kernel(const float* __restrict__ src,
                                unsigned short* __restrict__ dst, int n) {
  int i = blockIdx.x * blockDim.x + threadIdx.x;
  const int stride = gridDim.x * blockDim.x;
  for (; i < n; i += stride) dst[i] = f2bf(src[i]);
}

__global__ void bias_combine_kernel(const float* __restrict__ a,
                                    const float* __restrict__ b,
                                    float* __restrict__ o, int n) {
  int i = blockIdx.x * blockDim.x + threadIdx.x;
  if (i < n) o[i] = a[i] + b[i];
}

// ---------------------------------------------------------------------------
// One LSTM timestep: gates = [x_t | h] @ [W_ih | W_hh]^T + b, then pointwise.
// Wave layout: 4 waves/block, 128 blocks => 512 wave-tasks
//   task = (m_super in [0,8)) x (n_tile in [0,64))
// Each wave: 2 M-tiles (32 rows) x 1 N-tile (16 hidden cols) x 4 gates
//   => 8 f32 accumulator fragments, K-loop = 4 (F) + 32 (H) WMMA rounds.
// ---------------------------------------------------------------------------
__global__ __launch_bounds__(128)
void lstm_step_kernel(const unsigned short* __restrict__ x_bf,   // [B, T, F]
                      const unsigned short* __restrict__ Wih,    // [4H, F]
                      const unsigned short* __restrict__ Whh,    // [4H, H]
                      const float* __restrict__ bcomb,           // [4H]
                      const unsigned short* __restrict__ h_in,   // [B, H] bf16
                      unsigned short* __restrict__ h_out,        // [B, H] bf16
                      float* __restrict__ c,                     // [B, H] f32 (in-place)
                      int t)
{
  const int lane = threadIdx.x & 31;
  const int wave = threadIdx.x >> 5;
  const int task    = blockIdx.x * 4 + wave;   // 0..511
  const int m_super = task >> 6;               // 0..7
  const int n_tile  = task & 63;               // 0..63
  const int m0 = m_super * 32;
  const int n0 = n_tile * 16;

  const v8f z = {0.f, 0.f, 0.f, 0.f, 0.f, 0.f, 0.f, 0.f};
  v8f acc[2][4];
  #pragma unroll
  for (int i = 0; i < 2; ++i)
    #pragma unroll
    for (int g = 0; g < 4; ++g) acc[i][g] = z;

  // ---- phase 1: x_t @ W_ih^T (K = F = 128) ----
  const unsigned short* xbase = x_bf + (size_t)t * F_DIM;  // row stride T*F
  #pragma unroll
  for (int kc = 0; kc < F_DIM; kc += 32) {
    const v16bf a0 = load_frag_a(xbase, T_DIM * F_DIM, m0,      kc, lane);
    const v16bf a1 = load_frag_a(xbase, T_DIM * F_DIM, m0 + 16, kc, lane);
    #pragma unroll
    for (int g = 0; g < 4; ++g) {
      const v16bf b = load_frag_b(Wih, F_DIM, g * H_DIM + n0, kc, lane);
      acc[0][g] = __builtin_amdgcn_wmma_f32_16x16x32_bf16(false, a0, false, b,
                                                          (short)0, acc[0][g], false, false);
      acc[1][g] = __builtin_amdgcn_wmma_f32_16x16x32_bf16(false, a1, false, b,
                                                          (short)0, acc[1][g], false, false);
    }
  }

  // ---- phase 2: h @ W_hh^T (K = H = 1024) ----
  for (int kc = 0; kc < H_DIM; kc += 32) {
    const v16bf a0 = load_frag_a(h_in, H_DIM, m0,      kc, lane);
    const v16bf a1 = load_frag_a(h_in, H_DIM, m0 + 16, kc, lane);
    #pragma unroll
    for (int g = 0; g < 4; ++g) {
      const v16bf b = load_frag_b(Whh, H_DIM, g * H_DIM + n0, kc, lane);
      acc[0][g] = __builtin_amdgcn_wmma_f32_16x16x32_bf16(false, a0, false, b,
                                                          (short)0, acc[0][g], false, false);
      acc[1][g] = __builtin_amdgcn_wmma_f32_16x16x32_bf16(false, a1, false, b,
                                                          (short)0, acc[1][g], false, false);
    }
  }

  // ---- pointwise gates + state update ----
  // C/D layout: VGPR r, lanes 0-15 -> M=r, N=lane; lanes 16-31 -> M=r+8.
  const int ncol = n0 + (lane & 15);
  const int rhi  = (lane >> 4) << 3;
  const float bI = bcomb[ncol];
  const float bF = bcomb[H_DIM + ncol];
  const float bG = bcomb[2 * H_DIM + ncol];
  const float bO = bcomb[3 * H_DIM + ncol];

  #pragma unroll
  for (int mt = 0; mt < 2; ++mt) {
    #pragma unroll
    for (int r = 0; r < 8; ++r) {
      const int row  = m0 + mt * 16 + rhi + r;
      const size_t idx = (size_t)row * H_DIM + ncol;
      const float gi = sigmoidf_fast(acc[mt][0][r] + bI);
      const float gf = sigmoidf_fast(acc[mt][1][r] + bF);
      const float gg = tanhf_fast  (acc[mt][2][r] + bG);
      const float go = sigmoidf_fast(acc[mt][3][r] + bO);
      const float cn = gf * c[idx] + gi * gg;
      c[idx] = cn;
      h_out[idx] = f2bf(go * tanhf_fast(cn));
    }
  }
}

// ---------------------------------------------------------------------------
// Final FC: out = sigmoid(h) @ W_fc^T + b_fc   (M=256, N=128, K=1024)
// 128 wave-tasks = 16 M-tiles x 8 N-tiles; sigmoid applied on A-fragment.
// ---------------------------------------------------------------------------
__global__ __launch_bounds__(128)
void fc_kernel(const unsigned short* __restrict__ h_bf,   // [B, H] bf16
               const unsigned short* __restrict__ Wfc,    // [OUT, H] bf16
               const float* __restrict__ b_fc,            // [OUT]
               float* __restrict__ out)                   // [B, OUT] f32
{
  const int lane = threadIdx.x & 31;
  const int wave = threadIdx.x >> 5;
  const int task   = blockIdx.x * 4 + wave;  // 0..127
  const int m_tile = task >> 3;              // 0..15
  const int n_tile = task & 7;               // 0..7
  const int m0 = m_tile * 16;
  const int n0 = n_tile * 16;

  v8f acc = {0.f, 0.f, 0.f, 0.f, 0.f, 0.f, 0.f, 0.f};

  for (int kc = 0; kc < H_DIM; kc += 32) {
    FragU u;
    u.v = load_frag_a(h_bf, H_DIM, m0, kc, lane);
    #pragma unroll
    for (int e = 0; e < 16; ++e) u.s[e] = f2bf(sigmoidf_fast(bf2f(u.s[e])));
    const v16bf a = u.v;
    const v16bf b = load_frag_b(Wfc, H_DIM, n0, kc, lane);
    acc = __builtin_amdgcn_wmma_f32_16x16x32_bf16(false, a, false, b,
                                                  (short)0, acc, false, false);
  }

  const int ncol = n0 + (lane & 15);
  const int rhi  = (lane >> 4) << 3;
  const float bias = b_fc[ncol];
  #pragma unroll
  for (int r = 0; r < 8; ++r)
    out[(size_t)(m0 + rhi + r) * OUT_DIM + ncol] = acc[r] + bias;
}

// ---------------------------------------------------------------------------
extern "C" void kernel_launch(void* const* d_in, const int* in_sizes, int n_in,
                              void* d_out, int out_size, void* d_ws, size_t ws_size,
                              hipStream_t stream) {
  const float* x    = (const float*)d_in[0];  // [B, T, F]
  const float* h0   = (const float*)d_in[1];  // [B, H]
  const float* c0   = (const float*)d_in[2];  // [B, H]
  const float* W_ih = (const float*)d_in[3];  // [4H, F]
  const float* W_hh = (const float*)d_in[4];  // [4H, H]
  const float* b_ih = (const float*)d_in[5];  // [4H]
  const float* b_hh = (const float*)d_in[6];  // [4H]
  const float* W_fc = (const float*)d_in[7];  // [OUT, H]
  const float* b_fc = (const float*)d_in[8];  // [OUT]
  float* out = (float*)d_out;

  // Workspace carve-out (~28.6 MB total), 256B aligned slabs.
  char* w = (char*)d_ws;
  size_t off = 0;
  auto carve = [&](size_t bytes) -> void* {
    void* p = w + off;
    off = (off + bytes + 255) & ~(size_t)255;
    return p;
  };
  unsigned short* x_bf   = (unsigned short*)carve((size_t)B_DIM * T_DIM * F_DIM * 2);
  unsigned short* Wih_bf = (unsigned short*)carve((size_t)4 * H_DIM * F_DIM * 2);
  unsigned short* Whh_bf = (unsigned short*)carve((size_t)4 * H_DIM * H_DIM * 2);
  unsigned short* Wfc_bf = (unsigned short*)carve((size_t)OUT_DIM * H_DIM * 2);
  unsigned short* hbuf0  = (unsigned short*)carve((size_t)B_DIM * H_DIM * 2);
  unsigned short* hbuf1  = (unsigned short*)carve((size_t)B_DIM * H_DIM * 2);
  float*          cbuf   = (float*)carve((size_t)B_DIM * H_DIM * 4);
  float*          bcomb  = (float*)carve((size_t)4 * H_DIM * 4);

  // ---- prep: f32 -> bf16 conversions, bias combine, c init ----
  cvt_bf16_kernel<<<512, 256, 0, stream>>>(x,    x_bf,   B_DIM * T_DIM * F_DIM);
  cvt_bf16_kernel<<<256, 256, 0, stream>>>(W_ih, Wih_bf, 4 * H_DIM * F_DIM);
  cvt_bf16_kernel<<<512, 256, 0, stream>>>(W_hh, Whh_bf, 4 * H_DIM * H_DIM);
  cvt_bf16_kernel<<<64,  256, 0, stream>>>(W_fc, Wfc_bf, OUT_DIM * H_DIM);
  cvt_bf16_kernel<<<64,  256, 0, stream>>>(h0,   hbuf0,  B_DIM * H_DIM);
  bias_combine_kernel<<<16, 256, 0, stream>>>(b_ih, b_hh, bcomb, 4 * H_DIM);
  hipMemcpyAsync(cbuf, c0, (size_t)B_DIM * H_DIM * sizeof(float),
                 hipMemcpyDeviceToDevice, stream);

  // ---- sequential scan: one launch per timestep, h double-buffered ----
  for (int t = 0; t < T_DIM; ++t) {
    const unsigned short* hin  = (t & 1) ? hbuf1 : hbuf0;
    unsigned short*       hout = (t & 1) ? hbuf0 : hbuf1;
    lstm_step_kernel<<<128, 128, 0, stream>>>(x_bf, Wih_bf, Whh_bf, bcomb,
                                              hin, hout, cbuf, t);
  }

  // T=256 even: final h resides in hbuf0 (t=255 read hbuf1, wrote hbuf0)
  fc_kernel<<<32, 128, 0, stream>>>(hbuf0, Wfc_bf, b_fc, out);

  (void)in_sizes; (void)n_in; (void)out_size; (void)ws_size;
}
